// HSLCA_62543313764778
// MI455X (gfx1250) — compile-verified, matrix operand order checked by reading, not run on hardware
//
#include <hip/hip_runtime.h>
#include <cstdint>

typedef __attribute__((ext_vector_type(16))) __bf16 v16bf;
typedef __attribute__((ext_vector_type(8)))  float  v8f;
typedef __attribute__((ext_vector_type(4)))  unsigned int u32x4;
typedef __attribute__((ext_vector_type(8)))  int    i32x8;
typedef __attribute__((ext_vector_type(4)))  int    i32x4;

#define DIMC 256
#define NTOK 16384
#define NB   8
#define TM   64           // token rows per block in the GEMM-chain kernel

// ---------------- WMMA fragment helpers (16x16x32 bf16) ----------------
// A-matrix 16x32 bf16 layout (ISA 7.12.2): lane m = lane&15; lanes 0-15 hold
// K pairs (0,1)..(6,7) then (16,17)..(22,23); lanes 16-31 are +8 in K.
__device__ inline v16bf ldA(const __bf16* base, int ldstride, int row0, int k0, int lane) {
    v16bf f;
    const __bf16* p = base + (row0 + (lane & 15)) * ldstride + k0 + ((lane & 16) ? 8 : 0);
#pragma unroll
    for (int e = 0; e < 4; ++e) {
        f[2 * e]     = p[2 * e];
        f[2 * e + 1] = p[2 * e + 1];
        f[2 * e + 8] = p[2 * e + 16];
        f[2 * e + 9] = p[2 * e + 17];
    }
    return f;
}

// B fragments are pre-packed in lane order: tile*512 + lane*16 + e  (contiguous)
__device__ inline v16bf ldB(const __bf16* pk, int tile, int lane) {
    return *(const v16bf*)(pk + (size_t)tile * 512 + lane * 16);
}

__device__ inline v8f wmma_bf16(v16bf a, v16bf b, v8f c) {
    return __builtin_amdgcn_wmma_f32_16x16x32_bf16(false, a, false, b, (short)0, c, false, false);
}

// kmap used when PACKING B tiles (must mirror ldA's K mapping)
__device__ inline int kmap(int lane, int e) {
    return ((e < 8) ? e : e + 8) + ((lane & 16) ? 8 : 0);
}

// ---------------- Kernel W: pack q_w / out_w into WMMA B-format bf16 ----------------
// B[k=c][n=o] = W[o][c]; tiles: kt 0..7 (K=32 each), nt 0..15 (N=16 each)
__global__ __launch_bounds__(32) void hslca_packw(const float* __restrict__ q_w,
                                                  const float* __restrict__ out_w,
                                                  __bf16* __restrict__ qpk,
                                                  __bf16* __restrict__ opk) {
    int lane = threadIdx.x;
    int tile = blockIdx.x & 127;
    const float* src = (blockIdx.x < 128) ? q_w : out_w;
    __bf16* dst = (blockIdx.x < 128) ? qpk : opk;
    int kt = tile >> 4, nt = tile & 15;
    int n = nt * 16 + (lane & 15);
#pragma unroll
    for (int e = 0; e < 16; ++e) {
        int k = kt * 32 + kmap(lane, e);
        dst[(size_t)tile * 512 + lane * 16 + e] = (__bf16)src[n * DIMC + k];
    }
}

// ---------------- Kernel A: text summary -> K,V -> packed KV; zero colsum ----------------
__global__ __launch_bounds__(256) void hslca_prep(const float* __restrict__ text,
                                                  const float* __restrict__ sw,
                                                  const float* __restrict__ sb,
                                                  const float* __restrict__ k_w,
                                                  const float* __restrict__ k_b,
                                                  const float* __restrict__ v_w,
                                                  const float* __restrict__ v_b,
                                                  __bf16* __restrict__ kvpk,
                                                  float* __restrict__ colsum) {
    __shared__ float sc[128 * 4];
    __shared__ float ts[4 * DIMC];
    __shared__ float Km[4 * DIMC];
    __shared__ float Vm[4 * DIMC];
    __shared__ float red[8];
    int tid = threadIdx.x, b = blockIdx.x;
    const float* tx = text + (size_t)b * 128 * DIMC;
    colsum[b * DIMC + tid] = 0.f;

    if (tid < 128) {
        float a0 = sb[0], a1 = sb[1], a2 = sb[2], a3 = sb[3];
        for (int c = 0; c < DIMC; ++c) {
            float x = tx[tid * DIMC + c];
            a0 += x * sw[0 * DIMC + c]; a1 += x * sw[1 * DIMC + c];
            a2 += x * sw[2 * DIMC + c]; a3 += x * sw[3 * DIMC + c];
        }
        sc[tid * 4 + 0] = a0; sc[tid * 4 + 1] = a1; sc[tid * 4 + 2] = a2; sc[tid * 4 + 3] = a3;
    }
    __syncthreads();
    if (tid < 4) {
        float mx = -1e30f;
        for (int l = 0; l < 128; ++l) mx = fmaxf(mx, sc[l * 4 + tid]);
        float s = 0.f;
        for (int l = 0; l < 128; ++l) s += __expf(sc[l * 4 + tid] - mx);
        red[tid] = mx; red[4 + tid] = 1.f / s;
    }
    __syncthreads();
    if (tid < 128)
        for (int k = 0; k < 4; ++k) sc[tid * 4 + k] = __expf(sc[tid * 4 + k] - red[k]) * red[4 + k];
    __syncthreads();
    {   // text_summary[k][c]
        float a[4] = {0.f, 0.f, 0.f, 0.f};
        for (int l = 0; l < 128; ++l) {
            float x = tx[l * DIMC + tid];
            for (int k = 0; k < 4; ++k) a[k] += sc[l * 4 + k] * x;
        }
        for (int k = 0; k < 4; ++k) ts[k * DIMC + tid] = a[k];
    }
    __syncthreads();
    {   // K (phi) and V rows, o = tid
        int o = tid;
        for (int k = 0; k < 4; ++k) {
            float ak = k_b[o], av = v_b[o];
            for (int c = 0; c < DIMC; ++c) {
                float t = ts[k * DIMC + c];
                ak += t * k_w[o * DIMC + c];
                av += t * v_w[o * DIMC + c];
            }
            ak = (ak > 0.f) ? (ak + 1.f) : __expf(ak);   // elu+1
            Km[k * DIMC + o] = ak;
            Vm[k * DIMC + o] = av;
        }
    }
    __syncthreads();
    // KV[h][d][e] = sum_k K[k][h*64+d] * V[k][h*64+e], packed as WMMA B tiles:
    // tile = h*8 + kt*4 + ntl  (kt: K-tile of 32 over d, ntl: N-tile of 16 over e)
    for (int idx = tid; idx < 32 * 512; idx += 256) {
        int tile = idx >> 9, w = idx & 511, lane = w >> 4, e = w & 15;
        int h = tile >> 3, kt = (tile >> 2) & 1, ntl = tile & 3;
        int d  = kt * 32 + kmap(lane, e);
        int ee = ntl * 16 + (lane & 15);
        float acc = 0.f;
        for (int k = 0; k < 4; ++k) acc += Km[k * DIMC + h * 64 + d] * Vm[k * DIMC + h * 64 + ee];
        kvpk[(size_t)b * 32 * 512 + idx] = (__bf16)acc;
    }
}

// ---------------- Kernel B: fused GEMM chain (TDM + WMMA) ----------------
__global__ __launch_bounds__(256) void hslca_gemm_chain(const float* __restrict__ img,
                                                        const __bf16* __restrict__ qpk,
                                                        const __bf16* __restrict__ opk,
                                                        const __bf16* __restrict__ kvpk,
                                                        const float* __restrict__ q_b,
                                                        const float* __restrict__ out_b,
                                                        float* __restrict__ attn_out,
                                                        float* __restrict__ colsum) {
    __shared__ float  sImg[TM * DIMC];   // 64 KB TDM landing zone (f32)
    __shared__ __bf16 sA[TM * DIMC];     // 32 KB: img bf16, later attn bf16
    __shared__ __bf16 sQ[TM * DIMC];     // 32 KB: phi(Q) bf16
    __shared__ float  sCol[DIMC];

    const int tid  = threadIdx.x;
    const int lane = tid & 31;
    const int wave = tid >> 5;
    const int b = blockIdx.x >> 8;       // 256 tiles per batch
    const int t = blockIdx.x & 255;
    const size_t rowbase = ((size_t)b << 14) + (size_t)t * TM;

    sCol[tid] = 0.f;

    // --- Tensor Data Mover: DMA the 64x256 f32 tile into LDS ---
    if (tid < 32) {
        unsigned long long ga = (unsigned long long)(uintptr_t)(img + rowbase * DIMC);
        unsigned ldsOff = (unsigned)(uintptr_t)(void*)sImg;
        u32x4 g0 = {0u, 0u, 0u, 0u};
        g0[0] = 1u;                                               // count=1, user D#
        g0[1] = ldsOff;                                           // lds_addr
        g0[2] = (unsigned)(ga & 0xFFFFFFFFull);                   // global_addr[31:0]
        g0[3] = (unsigned)((ga >> 32) & 0x01FFFFFFull) | (2u << 30); // addr[56:32] | type=2
        i32x8 g1 = {0, 0, 0, 0, 0, 0, 0, 0};
        g1[0] = (int)(2u << 16);                  // data_size = 4B, no multicast
        g1[1] = (int)((unsigned)DIMC << 16);      // tensor_dim0[15:0] @ bits 63:48
        g1[2] = (int)((unsigned)TM << 16);        // tensor_dim1[15:0] @ bits 95:80
        g1[3] = (int)((unsigned)DIMC << 16);      // tile_dim0 @ bits 127:112
        g1[4] = TM;                               // tile_dim1 @ bits 143:128
        g1[5] = DIMC;                             // tensor_dim0_stride[31:0]
        i32x4 z4 = {0, 0, 0, 0};
        i32x8 z8 = {0, 0, 0, 0, 0, 0, 0, 0};
        __builtin_amdgcn_tensor_load_to_lds(g0, g1, z4, z4, z8, 0);
        __builtin_amdgcn_s_wait_tensorcnt(0);
    }
    __syncthreads();

    // f32 -> bf16 staging for WMMA A-side
    for (int i = tid; i < TM * DIMC; i += 256) sA[i] = (__bf16)sImg[i];
    __syncthreads();

    const int mt = wave >> 1;                 // 4 row tiles, 2 waves each
    const int nbase = (wave & 1) * 8;         // each wave covers 8 of 16 N-tiles
    const int mrow = mt * 16 + ((lane & 16) ? 8 : 0);
    const int ncol = lane & 15;

    // ---- stage 1: Q = phi(img @ q_w^T + q_b) ----
    v16bf afr[8];
#pragma unroll
    for (int kt = 0; kt < 8; ++kt) afr[kt] = ldA(sA, DIMC, mt * 16, kt * 32, lane);
#pragma unroll
    for (int j = 0; j < 8; ++j) {
        int nt = nbase + j;
        int n = nt * 16 + ncol;
        v8f acc = {};
#pragma unroll
        for (int kt = 0; kt < 8; ++kt)
            acc = wmma_bf16(afr[kt], ldB(qpk, kt * 16 + nt, lane), acc);
        float bias = q_b[n];
#pragma unroll
        for (int r = 0; r < 8; ++r) {
            float x = acc[r] + bias;
            x = (x > 0.f) ? (x + 1.f) : __expf(x);    // elu+1
            sQ[(mrow + r) * DIMC + n] = (__bf16)x;
        }
    }
    __syncthreads();

    // ---- stage 2: attn = Q @ KV  (block-diagonal per head, K=64) ----
#pragma unroll
    for (int j = 0; j < 8; ++j) {
        int nt = nbase + j;
        int h = nt >> 2, ntl = nt & 3;
        v8f acc = {};
#pragma unroll
        for (int kt = 0; kt < 2; ++kt) {
            v16bf a = ldA(sQ, DIMC, mt * 16, h * 64 + kt * 32, lane);
            acc = wmma_bf16(a, ldB(kvpk, b * 32 + h * 8 + kt * 4 + ntl, lane), acc);
        }
        int n = nt * 16 + ncol;
#pragma unroll
        for (int r = 0; r < 8; ++r) sA[(mrow + r) * DIMC + n] = (__bf16)acc[r];
    }
    __syncthreads();

    // ---- stage 3: out = attn @ out_w^T + out_b ----
#pragma unroll
    for (int kt = 0; kt < 8; ++kt) afr[kt] = ldA(sA, DIMC, mt * 16, kt * 32, lane);
#pragma unroll
    for (int j = 0; j < 8; ++j) {
        int nt = nbase + j;
        int n = nt * 16 + ncol;
        v8f acc = {};
#pragma unroll
        for (int kt = 0; kt < 8; ++kt)
            acc = wmma_bf16(afr[kt], ldB(opk, kt * 16 + nt, lane), acc);
        float bias = out_b[n];
        float s8 = 0.f;
#pragma unroll
        for (int r = 0; r < 8; ++r) {
            float x = acc[r] + bias;
            attn_out[(rowbase + mrow + r) * DIMC + n] = x;
            s8 += x;
        }
        atomicAdd(&sCol[n], s8);
    }
    __syncthreads();
    atomicAdd(&colsum[b * DIMC + tid], sCol[tid]);
}

// ---------------- B-splines (Cox–de Boor, k=3, grid 5) ----------------
__device__ inline void bspl8(float x, float* out) {
    float g[12];
#pragma unroll
    for (int i = 0; i < 12; ++i) g[i] = -1.0f + 0.4f * (float)(i - 3);
    float bb[11];
#pragma unroll
    for (int j = 0; j < 11; ++j) bb[j] = (x >= g[j] && x < g[j + 1]) ? 1.0f : 0.0f;
#pragma unroll
    for (int k = 1; k <= 3; ++k)
#pragma unroll
        for (int i = 0; i < 11; ++i) {
            if (i < 11 - k)
                bb[i] = (x - g[i]) / (g[i + k] - g[i]) * bb[i] +
                        (g[i + k + 1] - x) / (g[i + k + 1] - g[i + 1]) * bb[i + 1];
        }
#pragma unroll
    for (int i = 0; i < 8; ++i) out[i] = bb[i];
}

// ---------------- Kernel C: gate (layernorm + KAN1 + KAN2 -> alpha) ----------------
__global__ __launch_bounds__(256) void hslca_gate(const float* __restrict__ colsum,
                                                  const float* __restrict__ gg,
                                                  const float* __restrict__ gb,
                                                  const float* __restrict__ k1b,
                                                  const float* __restrict__ k1s,
                                                  const float* __restrict__ k2b,
                                                  const float* __restrict__ k2s,
                                                  float* __restrict__ alpha) {
    __shared__ float sx[DIMC];
    __shared__ float bs[DIMC * 8];
    __shared__ float h1[64];
    __shared__ float red[DIMC];
    int tid = threadIdx.x, b = blockIdx.x;
    float x = colsum[b * DIMC + tid] * (1.0f / (float)NTOK);
    red[tid] = x; __syncthreads();
    for (int s = 128; s > 0; s >>= 1) { if (tid < s) red[tid] += red[tid + s]; __syncthreads(); }
    float m = red[0] * (1.f / 256.f); __syncthreads();
    red[tid] = (x - m) * (x - m); __syncthreads();
    for (int s = 128; s > 0; s >>= 1) { if (tid < s) red[tid] += red[tid + s]; __syncthreads(); }
    float v = red[0] * (1.f / 256.f);
    x = (x - m) * rsqrtf(v + 1e-5f) * gg[tid] + gb[tid];

    sx[tid] = x / (1.f + __expf(-x));          // silu
    bspl8(x, &bs[tid * 8]);
    __syncthreads();
    if (tid < 64) {
        float acc = 0.f;
        for (int c = 0; c < DIMC; ++c) {
            acc += sx[c] * k1b[tid * DIMC + c];
            const float* sp = k1s + ((size_t)tid * DIMC + c) * 8;
            const float* bq = &bs[c * 8];
#pragma unroll
            for (int g = 0; g < 8; ++g) acc += bq[g] * sp[g];
        }
        h1[tid] = acc;
    }
    __syncthreads();
    if (tid < 64) {
        float y = h1[tid];
        sx[tid] = y / (1.f + __expf(-y));
        bspl8(y, &bs[tid * 8]);
    }
    __syncthreads();
    {
        float acc = 0.f;
        for (int i = 0; i < 64; ++i) {
            acc += sx[i] * k2b[tid * 64 + i];
            const float* sp = k2s + ((size_t)tid * 64 + i) * 8;
            const float* bq = &bs[i * 8];
#pragma unroll
            for (int g = 0; g < 8; ++g) acc += bq[g] * sp[g];
        }
        alpha[b * DIMC + tid] = 1.f / (1.f + __expf(-acc));
    }
}

// ---------------- Kernel D: fused residual + layernorm (wave32 per row) ----------------
__global__ __launch_bounds__(256) void hslca_fuse_ln(const float* __restrict__ img,
                                                     const float* __restrict__ attn,
                                                     const float* __restrict__ alpha,
                                                     const float* __restrict__ ng,
                                                     const float* __restrict__ nb,
                                                     float* __restrict__ out) {
    int tid = threadIdx.x, lane = tid & 31, wave = tid >> 5;
    size_t row = (size_t)blockIdx.x * 8 + wave;
    int b = (int)(row >> 14);
    const float* ip = img + row * DIMC + lane * 8;
    const float* ap = attn + row * DIMC + lane * 8;
    const float* al = alpha + b * DIMC + lane * 8;
    float f[8]; float s = 0.f, s2 = 0.f;
#pragma unroll
    for (int i = 0; i < 8; ++i) {
        float x = ip[i] + al[i] * ap[i];
        f[i] = x; s += x; s2 += x * x;
    }
#pragma unroll
    for (int o = 16; o > 0; o >>= 1) {
        s  += __shfl_xor(s, o, 32);
        s2 += __shfl_xor(s2, o, 32);
    }
    float m = s * (1.f / 256.f);
    float v = s2 * (1.f / 256.f) - m * m;
    float inv = rsqrtf(v + 1e-5f);
    float* op = out + row * DIMC + lane * 8;
    const float* gp = ng + lane * 8;
    const float* bp = nb + lane * 8;
#pragma unroll
    for (int i = 0; i < 8; ++i) op[i] = (f[i] - m) * inv * gp[i] + bp[i];
}

extern "C" void kernel_launch(void* const* d_in, const int* in_sizes, int n_in,
                              void* d_out, int out_size, void* d_ws, size_t ws_size,
                              hipStream_t stream) {
    const float* img   = (const float*)d_in[0];
    const float* text  = (const float*)d_in[1];
    const float* sw    = (const float*)d_in[2];
    const float* sb    = (const float*)d_in[3];
    const float* q_w   = (const float*)d_in[4];
    const float* q_b   = (const float*)d_in[5];
    const float* k_w   = (const float*)d_in[6];
    const float* k_b   = (const float*)d_in[7];
    const float* v_w   = (const float*)d_in[8];
    const float* v_b   = (const float*)d_in[9];
    const float* out_w = (const float*)d_in[10];
    const float* out_b = (const float*)d_in[11];
    const float* k1b   = (const float*)d_in[12];
    const float* k1s   = (const float*)d_in[13];
    const float* k2b   = (const float*)d_in[14];
    const float* k2s   = (const float*)d_in[15];
    const float* gg    = (const float*)d_in[16];
    const float* gb    = (const float*)d_in[17];
    const float* ng    = (const float*)d_in[18];
    const float* nb    = (const float*)d_in[19];
    float* out = (float*)d_out;

    char* ws = (char*)d_ws;
    float*  attn   = (float*)ws;   ws += (size_t)NB * NTOK * DIMC * sizeof(float);
    __bf16* qpk    = (__bf16*)ws;  ws += (size_t)128 * 512 * sizeof(__bf16);
    __bf16* opk    = (__bf16*)ws;  ws += (size_t)128 * 512 * sizeof(__bf16);
    __bf16* kvpk   = (__bf16*)ws;  ws += (size_t)NB * 32 * 512 * sizeof(__bf16);
    float*  colsum = (float*)ws;   ws += (size_t)NB * DIMC * sizeof(float);
    float*  alphap = (float*)ws;

    hslca_packw<<<256, 32, 0, stream>>>(q_w, out_w, qpk, opk);
    hslca_prep<<<NB, 256, 0, stream>>>(text, sw, sb, k_w, k_b, v_w, v_b, kvpk, colsum);
    hslca_gemm_chain<<<NB * (NTOK / TM), 256, 0, stream>>>(img, qpk, opk, kvpk, q_b, out_b, attn, colsum);
    hslca_gate<<<NB, 256, 0, stream>>>(colsum, gg, gb, k1b, k1s, k2b, k2s, alphap);
    hslca_fuse_ln<<<(NB * NTOK) / 8, 256, 0, stream>>>(img, attn, alphap, ng, nb, out);
}
